// reblurWithKernel_75935021793347
// MI455X (gfx1250) — matrix operand order repaired and checked by reading, not compiled, not run on hardware
//
#include <hip/hip_runtime.h>

typedef __attribute__((ext_vector_type(16))) __bf16 v16bf;
typedef __attribute__((ext_vector_type(8)))  float  v8f;

#define IMG_H   512
#define NT      128          // tiles per dimension (512/4)
#define KS      33
#define WIN     36           // window rows (and valid cols)
#define WROW    40           // padded window row stride == ZSTRIDE (cols 36..39 are zero)
#define KDIM    (WIN*WROW)   // 1440 = 45*32, K dimension, exactly chunked
#define NCHUNK  45
#define ZSTRIDE 40           // zero-framed kernel: 40 rows x 40 cols (bf16)
#define ZSIZE   (ZSTRIDE*ZSTRIDE)      // 1600 u16
#define WSIZE   (3*KDIM)               // 4320 u16
#define TLDS    (ZSIZE + WSIZE)        // 5920 u16 = 11840 B per tile (32B multiple)
#define WPW     4                      // waves (tiles) per workgroup

static __device__ __forceinline__ unsigned short f2bf(float f) {
    unsigned int u = __builtin_bit_cast(unsigned int, f);
    u = (u + 0x7FFFu + ((u >> 16) & 1u)) >> 16;   // round-to-nearest-even
    return (unsigned short)u;
}

__global__ __launch_bounds__(128) void reblur_wmma_kernel(
    const float* __restrict__ img,     // (3, 512, 512)
    const float* __restrict__ kern,    // (1089, 128, 128)
    float* __restrict__ out)           // (3, 512, 512)
{
    __shared__ __align__(32) unsigned short smem[WPW * TLDS];

    const int lane = threadIdx.x & 31;
    const int wave = threadIdx.x >> 5;
    const int tile = blockIdx.x * WPW + wave;
    const int h = tile >> 7;          // tile row
    const int w = tile & (NT - 1);    // tile col

    unsigned short* Z  = &smem[wave * TLDS];  // 40x40 zero-framed bf16 kernel
    unsigned short* Wn = Z + ZSIZE;           // 3 x (36x40) bf16 window, cols 36..39 zero

    // ---- stage 1: fill LDS (per-wave private region) ----
    for (int idx = lane; idx < ZSIZE; idx += 32) Z[idx] = 0;

    // per-tile 33x33 kernel: element (p,q) at kern[(p*33+q)*16384 + h*128 + w]
    const float* kt = kern + h * NT + w;
    for (int idx = lane; idx < KS * KS; idx += 32) {
        if (idx + 32 < KS * KS)
            __builtin_prefetch(&kt[(idx + 32) * (NT * NT)], 0, 1);
        float val = kt[idx * (NT * NT)];
        int p = idx / KS;
        int q = idx - p * KS;
        Z[(p + 3) * ZSTRIDE + (q + 3)] = f2bf(val);
    }

    // 36x40 edge-clamped window per channel -> bf16 (pad cols -> 0)
    const int y0 = h * 4 - 16, x0 = w * 4 - 16;
    for (int idx = lane; idx < 3 * KDIM; idx += 32) {
        int c   = idx / KDIM;
        int rem = idx - c * KDIM;
        int u   = rem / WROW;
        int v   = rem - u * WROW;
        unsigned short val = 0;
        if (v < WIN) {
            int y = y0 + u; y = (y < 0) ? 0 : (y > IMG_H - 1 ? IMG_H - 1 : y);
            int x = x0 + v; x = (x < 0) ? 0 : (x > IMG_H - 1 ? IMG_H - 1 : x);
            val = f2bf(img[(c * IMG_H + y) * IMG_H + x]);
        }
        Wn[c * KDIM + rem] = val;
    }

    __syncthreads();

    // ---- stage 2: 16x3x1440 GEMM via 45 x v_wmma_f32_16x16x32_bf16 ----
    // k = u*40 + v  ==> A address = (u-i+3)*40 + (v-j+3) = k + abase : purely linear.
    const int laneHalf = lane >> 4;        // 0: K low half, 1: K high half
    const int m  = lane & 15;              // A row / D column index base
    const int ai = m >> 2, aj = m & 3;     // output (i,j) inside the 4x4 tile
    const int abase = (3 - ai) * ZSTRIDE + (3 - aj);
    const int bcol = (m < 3) ? m : 0;      // unused columns read channel 0 (harmless)
    const unsigned short* wcol = Wn + bcol * KDIM;
    const unsigned short* arow = Z + abase;

    v8f acc = {};

    #pragma unroll 3
    for (int ch = 0; ch < NCHUNK; ++ch) {
        const int kb0 = ch * 32 + laneHalf * 16;
        v16bf a;
        #pragma unroll
        for (int t = 0; t < 8; ++t) {      // elements 0..7 : K = kb0 + t (contiguous)
            a[t] = __builtin_bit_cast(__bf16, arow[kb0 + t]);
        }
        #pragma unroll
        for (int t = 0; t < 8; ++t) {      // elements 8..15 : K = kb0 + 8 + t (contiguous)
            a[8 + t] = __builtin_bit_cast(__bf16, arow[kb0 + 8 + t]);
        }
        // B: lane half selects K..K+15 vs K+16..K+31; 32B-aligned LDS vector load
        v16bf b = *(const v16bf*)(wcol + ch * 32 + laneHalf * 16);

        acc = __builtin_amdgcn_wmma_f32_16x16x32_bf16(
                  false, a, false, b, (short)0, acc, false, false);
    }

    // ---- stage 3: store D. VGPR r: M = r + 8*laneHalf, N = lane&15 = channel ----
    if (m < 3) {
        const int c = m;
        #pragma unroll
        for (int r = 0; r < 8; ++r) {
            int mm = r + laneHalf * 8;
            int y = h * 4 + (mm >> 2);
            int x = w * 4 + (mm & 3);
            out[(c * IMG_H + y) * IMG_H + x] = acc[r];
        }
    }
}

extern "C" void kernel_launch(void* const* d_in, const int* in_sizes, int n_in,
                              void* d_out, int out_size, void* d_ws, size_t ws_size,
                              hipStream_t stream) {
    const float* img  = (const float*)d_in[0];   // (1,3,512,512) fp32
    const float* kern = (const float*)d_in[1];   // (1,1089,128,128) fp32
    float* out = (float*)d_out;                  // (1,3,512,512) fp32

    const int numTiles = NT * NT;                // 16384
    dim3 grid(numTiles / WPW);                   // 4096 blocks
    dim3 block(32 * WPW);                        // 128 threads = 4 waves, 1 tile/wave
    reblur_wmma_kernel<<<grid, block, 0, stream>>>(img, kern, out);
}
